// ASAPooling_54219667145507
// MI455X (gfx1250) — matrix operand order repaired and checked by reading, not compiled, not run on hardware
//
#include <hip/hip_runtime.h>
#include <hip/hip_bf16.h>
#include <math.h>

#define NN 8192
#define CC 256
#define KSEL 4096
#define NEG_SLOPE 0.2f
#define ORD_NEG_INF 0x807FFFFF

typedef __attribute__((ext_vector_type(16))) __bf16 v16bf;
typedef __attribute__((ext_vector_type(8)))  float  v8f;

// ---------- helpers ----------
__device__ __forceinline__ int f2ord(float f) {
  int i = __float_as_int(f);
  return i ^ ((i >> 31) & 0x7fffffff);
}
__device__ __forceinline__ float ord2f(int i) {
  return __int_as_float(i ^ ((i >> 31) & 0x7fffffff));
}
__device__ __forceinline__ unsigned short f2bf(float f) {
  unsigned u = __float_as_uint(f);
  unsigned r = ((u >> 16) & 1u) + 0x7fffu;
  return (unsigned short)((u + r) >> 16);
}
__device__ __forceinline__ float wave_sum(float v) {
  for (int o = 16; o > 0; o >>= 1) v += __shfl_down(v, o, 32);
  return v;
}
__device__ __forceinline__ void fetch_edge(int e, const int* row, const int* col,
                                           const float* ew, int E,
                                           int& r, int& c, float& w) {
  if (e < E) { r = row[e]; c = col[e]; w = ew ? ew[e] : 0.f; }
  else       { r = e - E; c = e - E; w = 1.f; }
}

// ---------- fills ----------
__global__ void fill_i32_kernel(int* p, int n, int v) {
  for (int i = blockIdx.x * 256 + threadIdx.x; i < n; i += gridDim.x * 256) p[i] = v;
}
__global__ void fill_f32_kernel(float* p, long long n, float v) {
  long long stride = (long long)gridDim.x * 256;
  for (long long i = (long long)blockIdx.x * 256 + threadIdx.x; i < n; i += stride) p[i] = v;
}

// ---------- effective attention weight: w_eff = lin_w^T @ att_w[:C], b_eff = lin_b . att_w[:C] ----------
__global__ void prep_weff_kernel(const float* lin_w, const float* lin_b,
                                 const float* att_w, float* w_eff, float* b_eff) {
  int c = threadIdx.x;
  float s = 0.f;
  for (int o = 0; o < CC; ++o) s += lin_w[o * CC + c] * att_w[o];
  w_eff[c] = s;
  if (c == 0) {
    float b = 0.f;
    for (int o = 0; o < CC; ++o) b += lin_b[o] * att_w[o];
    *b_eff = b;
  }
}

// ---------- segment max of x[r] into target c (order-int atomics) ----------
__global__ void segmax_kernel(const int* row, const int* col, const float* x,
                              int* xq_ord, int E, int Etot) {
  int total = Etot * CC;
  int i = blockIdx.x * 256 + threadIdx.x;
  if (i >= total) return;
  int e = i / CC, ch = i - e * CC;
  int r, c; float w;
  fetch_edge(e, row, col, nullptr, E, r, c, w);
  atomicMax(&xq_ord[c * CC + ch], f2ord(x[r * CC + ch]));
}

// ---------- per-node a_q, a_j (wave-per-node dot products) ----------
__global__ void node_scal_kernel(const int* xq_ord, const float* x, const float* w_eff,
                                 const float* b_eff, const float* att_w,
                                 float* aq, float* aj) {
  int wv = threadIdx.x >> 5, lane = threadIdx.x & 31;
  int i = blockIdx.x * 8 + wv;
  if (i >= NN) return;
  float sq = 0.f, sj = 0.f;
  for (int ch = lane; ch < CC; ch += 32) {
    sq += ord2f(xq_ord[i * CC + ch]) * w_eff[ch];
    sj += x[i * CC + ch] * att_w[CC + ch];
  }
  sq = wave_sum(sq); sj = wave_sum(sj);
  if (lane == 0) { aq[i] = sq + *b_eff; aj[i] = sj; }
}

// ---------- edge raw score + segment max ----------
__global__ void edge_raw_kernel(const int* row, const int* col, const float* aq,
                                const float* aj, const float* att_b,
                                float* score, int* m_ord, int E, int Etot) {
  int e = blockIdx.x * 256 + threadIdx.x;
  if (e >= Etot) return;
  int r, c; float w;
  fetch_edge(e, row, col, nullptr, E, r, c, w);
  float v = aq[c] + aj[r] + att_b[0];
  v = v > 0.f ? v : NEG_SLOPE * v;
  score[e] = v;
  atomicMax(&m_ord[c], f2ord(v));
}

// ---------- edge exp + segment sum + degree ----------
__global__ void edge_exp_kernel(const int* row, const int* col, const int* m_ord,
                                float* score, float* ssum, float* deg, int E, int Etot) {
  int e = blockIdx.x * 256 + threadIdx.x;
  if (e >= Etot) return;
  int r, c; float w;
  fetch_edge(e, row, col, nullptr, E, r, c, w);
  float ee = expf(score[e] - ord2f(m_ord[c]));
  score[e] = ee;
  atomicAdd(&ssum[c], ee);
  atomicAdd(&deg[c], 1.0f);
}

// ---------- normalize ----------
__global__ void edge_norm_kernel(const int* row, const int* col, const float* ssum,
                                 float* score, int E, int Etot) {
  int e = blockIdx.x * 256 + threadIdx.x;
  if (e >= Etot) return;
  int r, c; float w;
  fetch_edge(e, row, col, nullptr, E, r, c, w);
  score[e] = score[e] / ssum[c];
}

// ---------- x_new scatter: x_new[c,:] += x[r,:] * score ----------
__global__ void xnew_scatter_kernel(const int* row, const int* col, const float* x,
                                    const float* score, float* xnew, int E, int Etot) {
  int total = Etot * CC;
  int i = blockIdx.x * 256 + threadIdx.x;
  if (i >= total) return;
  int e = i / CC, ch = i - e * CC;
  int r, c; float w;
  fetch_edge(e, row, col, nullptr, E, r, c, w);
  atomicAdd(&xnew[c * CC + ch], x[r * CC + ch] * score[e]);
}

// ---------- per-node a, fitbase = c3 - deg*b ----------
__global__ void node_abc_kernel(const float* xnew, const float* le1_w, const float* le1_b,
                                const float* le2_w, const float* le3_w, const float* le3_b,
                                const float* deg, float* a_node, float* fitbase) {
  int wv = threadIdx.x >> 5, lane = threadIdx.x & 31;
  int i = blockIdx.x * 8 + wv;
  if (i >= NN) return;
  float s1 = 0.f, s2 = 0.f, s3 = 0.f;
  for (int ch = lane; ch < CC; ch += 32) {
    float xv = xnew[i * CC + ch];
    s1 += xv * le1_w[ch];
    s2 += xv * le2_w[ch];
    s3 += xv * le3_w[ch];
  }
  s1 = wave_sum(s1); s2 = wave_sum(s2); s3 = wave_sum(s3);
  if (lane == 0) {
    a_node[i] = s1 + le1_b[0];
    fitbase[i] = (s3 + le3_b[0]) - deg[i] * s2;
  }
}

// ---------- agg = segment_sum(a[r], c) ----------
__global__ void edge_agg_kernel(const int* row, const int* col, const float* a_node,
                                float* aggfit, int E, int Etot) {
  int e = blockIdx.x * 256 + threadIdx.x;
  if (e >= Etot) return;
  int r, c; float w;
  fetch_edge(e, row, col, nullptr, E, r, c, w);
  atomicAdd(&aggfit[c], a_node[r]);
}

// ---------- fitness = sigmoid(agg + fitbase) ----------
__global__ void node_fit_kernel(const float* aggfit, const float* fitbase, float* fitness) {
  int i = blockIdx.x * 256 + threadIdx.x;
  if (i >= NN) return;
  float v = aggfit[i] + fitbase[i];
  fitness[i] = 1.0f / (1.0f + expf(-v));
}

// ---------- top-k via one-block bitonic sort (desc by val, asc by idx on ties) ----------
__global__ __launch_bounds__(1024) void topk_kernel(const float* fit, const int* batch,
                                                    float* fit_top, int* perm,
                                                    float* out_batch, float* out_perm) {
  __shared__ float sv[NN];
  __shared__ int si[NN];
  for (int i = threadIdx.x; i < NN; i += 1024) { sv[i] = fit[i]; si[i] = i; }
  __syncthreads();
  for (int k = 2; k <= NN; k <<= 1) {
    for (int j = k >> 1; j > 0; j >>= 1) {
      for (int t = threadIdx.x; t < NN; t += 1024) {
        int l = t ^ j;
        if (l > t) {
          bool up = ((t & k) == 0);
          float va = sv[t], vb = sv[l];
          int ia = si[t], ib = si[l];
          bool aFirst = (va > vb) || (va == vb && ia < ib);
          bool doswap = up ? !aFirst : aFirst;
          if (doswap) { sv[t] = vb; sv[l] = va; si[t] = ib; si[l] = ia; }
        }
      }
      __syncthreads();
    }
  }
  for (int t = threadIdx.x; t < KSEL; t += 1024) {
    perm[t] = si[t];
    fit_top[t] = sv[t];
    out_perm[t] = (float)si[t];
    out_batch[t] = (float)batch[si[t]];
  }
}

// ---------- invsel scatter ----------
__global__ void sel_kernel(const int* perm, int* invsel) {
  int j = blockIdx.x * 256 + threadIdx.x;
  if (j < KSEL) invsel[perm[j]] = j;
}

// ---------- x_out = x_new[perm] * fit_top ----------
__global__ void xout_kernel(const int* perm, const float* fit_top, const float* xnew,
                            float* xout) {
  int i = blockIdx.x * 256 + threadIdx.x;
  if (i >= KSEL * CC) return;
  int j = i / CC, ch = i - j * CC;
  xout[i] = xnew[perm[j] * CC + ch] * fit_top[j];
}

// ---------- S_sel scatter: S_sel[r, invsel[c]] += score ----------
__global__ void ssel_scatter_kernel(const int* row, const int* col, const float* score,
                                    const int* invsel, float* Ssel, int E, int Etot) {
  int e = blockIdx.x * 256 + threadIdx.x;
  if (e >= Etot) return;
  int r, c; float w;
  fetch_edge(e, row, col, nullptr, E, r, c, w);
  int jj = invsel[c];
  if (jj >= 0) atomicAdd(&Ssel[(size_t)r * KSEL + jj], score[e]);
}

// ---------- CSR build ----------
__global__ void csr_count_kernel(const int* row, const int* col, int* cnt, int E, int Etot) {
  int e = blockIdx.x * 256 + threadIdx.x;
  if (e >= Etot) return;
  int r, c; float w;
  fetch_edge(e, row, col, nullptr, E, r, c, w);
  atomicAdd(&cnt[r], 1);
}
__global__ __launch_bounds__(1024) void scan_kernel(const int* cnt, int* row_off, int* row_cur) {
  __shared__ int s[NN];
  int tid = threadIdx.x;
  for (int i = tid; i < NN; i += 1024) s[i] = cnt[i];
  __syncthreads();
  for (int off = 1; off < NN; off <<= 1) {
    int vals[8];
    for (int q = 0; q < 8; ++q) {
      int i = tid + q * 1024;
      vals[q] = (i >= off) ? s[i - off] : 0;
    }
    __syncthreads();
    for (int q = 0; q < 8; ++q) s[tid + q * 1024] += vals[q];
    __syncthreads();
  }
  if (tid == 0) row_off[0] = 0;
  for (int i = tid; i < NN; i += 1024) {
    row_off[i + 1] = s[i];
    row_cur[i] = (i == 0) ? 0 : s[i - 1];
  }
}
__global__ void csr_fill_kernel(const int* row, const int* col, const float* ew,
                                int* row_cur, int* csr_v, float* csr_w, int E, int Etot) {
  int e = blockIdx.x * 256 + threadIdx.x;
  if (e >= Etot) return;
  int r, c; float w;
  fetch_edge(e, row, col, ew, E, r, c, w);
  int p = atomicAdd(&row_cur[r], 1);
  csr_v[p] = c;
  csr_w[p] = w;
}

// ---------- SpMM: M = A_sparse @ S_sel, written as bf16. Block per row u. ----------
__global__ __launch_bounds__(256) void spmm_kernel(const int* row_off, const int* csr_v,
                                                   const float* csr_w, const float* Ssel,
                                                   unsigned short* Mb) {
  int u = blockIdx.x, t = threadIdx.x;
  float acc[16];
#pragma unroll
  for (int q = 0; q < 16; ++q) acc[q] = 0.f;
  int pb = row_off[u], pe = row_off[u + 1];
  for (int p = pb; p < pe; ++p) {
    int v = csr_v[p];
    float w = csr_w[p];
    const float* sr = Ssel + (size_t)v * KSEL;
#pragma unroll
    for (int q = 0; q < 16; ++q) acc[q] += w * sr[t + 256 * q];
  }
  size_t base = (size_t)u * KSEL;
#pragma unroll
  for (int q = 0; q < 16; ++q) Mb[base + t + 256 * q] = f2bf(acc[q]);
}

// ---------- transpose + bf16 convert: STb[j, v] = bf16(Ssel[v, j]) ----------
__global__ void transpose_kernel(const float* Ssel, unsigned short* STb) {
  __shared__ float tile[32][33];
  int j0 = blockIdx.x * 32, v0 = blockIdx.y * 32;
  int tx = threadIdx.x, ty = threadIdx.y;
  for (int q = 0; q < 4; ++q) {
    int vy = ty + q * 8;
    tile[vy][tx] = Ssel[(size_t)(v0 + vy) * KSEL + j0 + tx];
  }
  __syncthreads();
  for (int q = 0; q < 4; ++q) {
    int jy = ty + q * 8;
    STb[(size_t)(j0 + jy) * NN + v0 + tx] = f2bf(tile[tx][jy]);
  }
}

// ---------- WMMA GEMM: A2 = STb(4096x8192) @ Mb(8192x4096), f32 acc, zero diagonal ----------
// A tile staged with GLOBAL_LOAD_ASYNC_TO_LDS_B128 (ASYNCcnt path), B tile staged
// transposed via VALU+ds_store so each lane's B fragment is two contiguous b128 LDS loads.
__global__ __launch_bounds__(256) void gemm_kernel(const unsigned short* STb,
                                                   const unsigned short* Mb, float* A2) {
  __shared__ __align__(16) unsigned short As[128 * 32];   // [i_local][k]
  __shared__ __align__(16) unsigned short Bs[128 * 32];   // [n_local][k] (transposed stage)
  int tid = threadIdx.x;
  int i0 = blockIdx.y * 128, j0 = blockIdx.x * 128;
  int wv = tid >> 5, lane = tid & 31;
  int wr = wv & 1, wc = wv >> 1;       // 2x4 wave grid -> 64x32 per wave
  int m = lane & 15, hl = lane >> 4;

  v8f acc[4][2];
#pragma unroll
  for (int mi = 0; mi < 4; ++mi)
#pragma unroll
    for (int ni = 0; ni < 2; ++ni)
      acc[mi][ni] = (v8f){0.f, 0.f, 0.f, 0.f, 0.f, 0.f, 0.f, 0.f};

  for (int kk = 0; kk < NN; kk += 32) {
    // stage A: 128x32 bf16 row-major copy -> async global->LDS (tracked by ASYNCcnt)
#pragma unroll
    for (int s = 0; s < 2; ++s) {
      int ch = tid * 2 + s;
      int il = ch >> 2, kl = (ch & 3) * 8;
      const unsigned short* gp = &STb[(size_t)(i0 + il) * NN + kk + kl];
      unsigned lo = (unsigned)(size_t)&As[il * 32 + kl];
      asm volatile("global_load_async_to_lds_b128 %0, %1, off"
                   :: "v"(lo), "v"(gp) : "memory");
    }
    // stage B transposed into [n][k]
#pragma unroll
    for (int s = 0; s < 2; ++s) {
      int ch = tid * 2 + s;
      int kl = ch >> 4, n0 = (ch & 15) * 8;
      uint4 g = *(const uint4*)&Mb[(size_t)(kk + kl) * KSEL + j0 + n0];
      const unsigned short* gs = (const unsigned short*)&g;
#pragma unroll
      for (int w2 = 0; w2 < 8; ++w2) Bs[(n0 + w2) * 32 + kl] = gs[w2];
    }
    // prefetch next k-tile into cache (global_prefetch_b8)
    if (kk + 32 < NN) {
      int ch = tid * 2;
      __builtin_prefetch(&STb[(size_t)(i0 + (ch >> 2)) * NN + kk + 32 + ((ch & 3) * 8)], 0, 0);
      __builtin_prefetch(&Mb[(size_t)(kk + 32 + (ch >> 4)) * KSEL + j0 + ((ch & 15) * 8)], 0, 0);
    }
    // wave-local async completion, then workgroup-wide visibility
    asm volatile("s_wait_asynccnt 0x0" ::: "memory");
    __syncthreads();

    union Frag { v16bf v; uint4 q[2]; } fa[4], fb[2];
#pragma unroll
    for (int mi = 0; mi < 4; ++mi) {
      int rl = wr * 64 + mi * 16 + m;
      // wave32 16-bit A layout: lane<16 holds k in [8h..8h+7] U [16+8h..23+8h]
      fa[mi].q[0] = *(const uint4*)&As[rl * 32 + hl * 8];
      fa[mi].q[1] = *(const uint4*)&As[rl * 32 + 16 + hl * 8];
    }
#pragma unroll
    for (int ni = 0; ni < 2; ++ni) {
      int nl = wc * 32 + ni * 16 + m;
      // B layout: lane holds k in [8h..8h+7] U [16+8h..23+8h] of its column
      fb[ni].q[0] = *(const uint4*)&Bs[nl * 32 + hl * 8];
      fb[ni].q[1] = *(const uint4*)&Bs[nl * 32 + 16 + hl * 8];
    }
#pragma unroll
    for (int mi = 0; mi < 4; ++mi)
#pragma unroll
      for (int ni = 0; ni < 2; ++ni)
        acc[mi][ni] = __builtin_amdgcn_wmma_f32_16x16x32_bf16(
            false, fa[mi].v, false, fb[ni].v, (short)0, acc[mi][ni], false, false);
    __syncthreads();
  }

  // epilogue (C/D layout: lane<16 -> rows g, lane>=16 -> rows g+8; col = lane%16)
#pragma unroll
  for (int mi = 0; mi < 4; ++mi)
#pragma unroll
    for (int ni = 0; ni < 2; ++ni) {
      int jj = j0 + wc * 32 + ni * 16 + m;
#pragma unroll
      for (int g = 0; g < 8; ++g) {
        int ii = i0 + wr * 64 + mi * 16 + g + hl * 8;
        float v = acc[mi][ni][g];
        A2[(size_t)ii * KSEL + jj] = (ii == jj) ? 0.f : v;
      }
    }
}

// ============================ host launcher ============================
extern "C" void kernel_launch(void* const* d_in, const int* in_sizes, int n_in,
                              void* d_out, int out_size, void* d_ws, size_t ws_size,
                              hipStream_t stream) {
  (void)n_in; (void)out_size; (void)ws_size;
  const float* x      = (const float*)d_in[0];
  const int*   eidx   = (const int*)d_in[1];
  const float* ew     = (const float*)d_in[2];
  const int*   batch  = (const int*)d_in[3];
  const float* lin_w  = (const float*)d_in[4];
  const float* lin_b  = (const float*)d_in[5];
  const float* att_w  = (const float*)d_in[6];
  const float* att_b  = (const float*)d_in[7];
  const float* le1_w  = (const float*)d_in[8];
  const float* le1_b  = (const float*)d_in[9];
  const float* le2_w  = (const float*)d_in[10];
  const float* le3_w  = (const float*)d_in[11];
  const float* le3_b  = (const float*)d_in[12];

  const int E = in_sizes[2];
  const int Etot = E + NN;
  const int* row = eidx;
  const int* col = eidx + E;

  // output layout: x_out[K*C], A2[K*K], batch_out[K], perm[K]
  float* out      = (float*)d_out;
  float* out_x    = out;
  float* out_A2   = out + (size_t)KSEL * CC;
  float* out_bat  = out + (size_t)KSEL * CC + (size_t)KSEL * KSEL;
  float* out_perm = out_bat + KSEL;

  // workspace bump allocator
  char* ws = (char*)d_ws;
  size_t off = 0;
  auto alloc = [&](size_t bytes) -> char* {
    char* p = ws + off;
    off = (off + bytes + 255) & ~(size_t)255;
    return p;
  };
  int*   xq_ord  = (int*)  alloc((size_t)NN * CC * 4);
  float* xnew    = (float*)alloc((size_t)NN * CC * 4);
  float* w_eff   = (float*)alloc(CC * 4);
  float* b_eff   = (float*)alloc(256);
  float* aq      = (float*)alloc(NN * 4);
  float* aj      = (float*)alloc(NN * 4);
  int*   m_ord   = (int*)  alloc(NN * 4);
  float* ssum    = (float*)alloc(NN * 4);
  float* deg     = (float*)alloc(NN * 4);
  float* a_node  = (float*)alloc(NN * 4);
  float* fitbase = (float*)alloc(NN * 4);
  float* aggfit  = (float*)alloc(NN * 4);
  float* fitness = (float*)alloc(NN * 4);
  float* score   = (float*)alloc((size_t)Etot * 4);
  int*   perm    = (int*)  alloc(KSEL * 4);
  float* fit_top = (float*)alloc(KSEL * 4);
  int*   invsel  = (int*)  alloc(NN * 4);
  int*   cnt     = (int*)  alloc(NN * 4);
  int*   row_off = (int*)  alloc((NN + 1) * 4);
  int*   row_cur = (int*)  alloc(NN * 4);
  int*   csr_v   = (int*)  alloc((size_t)Etot * 4);
  float* csr_w   = (float*)alloc((size_t)Etot * 4);
  float* Ssel    = (float*)alloc((size_t)NN * KSEL * 4);
  unsigned short* Mb  = (unsigned short*)alloc((size_t)NN * KSEL * 2);
  unsigned short* STb = (unsigned short*)alloc((size_t)KSEL * NN * 2);

  const int T = 256;
  const int FB = 8192;  // fill blocks (grid-stride)

  // init
  fill_i32_kernel<<<FB, T, 0, stream>>>(xq_ord, NN * CC, ORD_NEG_INF);
  fill_i32_kernel<<<FB, T, 0, stream>>>(m_ord, NN, ORD_NEG_INF);
  fill_i32_kernel<<<FB, T, 0, stream>>>(invsel, NN, -1);
  fill_i32_kernel<<<FB, T, 0, stream>>>(cnt, NN, 0);
  fill_f32_kernel<<<FB, T, 0, stream>>>(ssum, NN, 0.f);
  fill_f32_kernel<<<FB, T, 0, stream>>>(deg, NN, 0.f);
  fill_f32_kernel<<<FB, T, 0, stream>>>(aggfit, NN, 0.f);
  fill_f32_kernel<<<FB, T, 0, stream>>>(xnew, (long long)NN * CC, 0.f);
  fill_f32_kernel<<<FB, T, 0, stream>>>(Ssel, (long long)NN * KSEL, 0.f);

  prep_weff_kernel<<<1, T, 0, stream>>>(lin_w, lin_b, att_w, w_eff, b_eff);

  int gEC = (Etot * CC + T - 1) / T;
  int gE  = (Etot + T - 1) / T;
  segmax_kernel<<<gEC, T, 0, stream>>>(row, col, x, xq_ord, E, Etot);
  node_scal_kernel<<<NN / 8, T, 0, stream>>>(xq_ord, x, w_eff, b_eff, att_w, aq, aj);
  edge_raw_kernel<<<gE, T, 0, stream>>>(row, col, aq, aj, att_b, score, m_ord, E, Etot);
  edge_exp_kernel<<<gE, T, 0, stream>>>(row, col, m_ord, score, ssum, deg, E, Etot);
  edge_norm_kernel<<<gE, T, 0, stream>>>(row, col, ssum, score, E, Etot);
  xnew_scatter_kernel<<<gEC, T, 0, stream>>>(row, col, x, score, xnew, E, Etot);
  node_abc_kernel<<<NN / 8, T, 0, stream>>>(xnew, le1_w, le1_b, le2_w, le3_w, le3_b,
                                            deg, a_node, fitbase);
  edge_agg_kernel<<<gE, T, 0, stream>>>(row, col, a_node, aggfit, E, Etot);
  node_fit_kernel<<<NN / T, T, 0, stream>>>(aggfit, fitbase, fitness);

  topk_kernel<<<1, 1024, 0, stream>>>(fitness, batch, fit_top, perm, out_bat, out_perm);
  sel_kernel<<<(KSEL + T - 1) / T, T, 0, stream>>>(perm, invsel);
  xout_kernel<<<(KSEL * CC + T - 1) / T, T, 0, stream>>>(perm, fit_top, xnew, out_x);

  ssel_scatter_kernel<<<gE, T, 0, stream>>>(row, col, score, invsel, Ssel, E, Etot);

  csr_count_kernel<<<gE, T, 0, stream>>>(row, col, cnt, E, Etot);
  scan_kernel<<<1, 1024, 0, stream>>>(cnt, row_off, row_cur);
  csr_fill_kernel<<<gE, T, 0, stream>>>(row, col, ew, row_cur, csr_v, csr_w, E, Etot);

  spmm_kernel<<<NN, T, 0, stream>>>(row_off, csr_v, csr_w, Ssel, Mb);
  transpose_kernel<<<dim3(KSEL / 32, NN / 32), dim3(32, 8), 0, stream>>>(Ssel, STb);
  gemm_kernel<<<dim3(KSEL / 128, KSEL / 128), T, 0, stream>>>(STb, Mb, out_A2);
}
